// Decoder_17738214932852
// MI455X (gfx1250) — matrix operand order repaired and checked by reading, not compiled, not run on hardware
//
#include <hip/hip_runtime.h>
#include <hip/hip_bf16.h>
#include <math.h>

// ---- problem constants (match reference) ----
#define NB   128      // batch
#define LL   256      // encoder length
#define HH   512      // hidden
#define VV   64       // vocab
#define TT   20       // steps
#define G4H  2048     // 4*H
#define K0   1536     // 2H + H (lstm0 fused K)
#define K1   1024     // H + H   (lstm1 fused K)

typedef __attribute__((ext_vector_type(16))) _Float16 v16h;
typedef __attribute__((ext_vector_type(8)))  _Float16 h8;
typedef __attribute__((ext_vector_type(8)))  float    v8f;

// ---------------- WMMA fragment loaders ----------------
// A fragment (16x32 f16, MxK), row-major source, lda in elements.
// ISA layout: lane l (m = l&15, hi = l>>4):
//   elems 0..7  -> k = k0 + hi*8 + i
//   elems 8..15 -> k = k0 + 16 + hi*8 + (i-8)
__device__ inline v16h load_frag_a(const _Float16* __restrict__ base, int lda,
                                   int m0, int k0) {
    int lane = threadIdx.x & 31;
    int r = lane & 15, hi = (lane >> 4) & 1;
    const _Float16* p = base + (size_t)(m0 + r) * lda + k0 + hi * 8;
    h8 lo = *(const h8*)p;
    h8 hv = *(const h8*)(p + 16);
    v16h a;
#pragma unroll
    for (int i = 0; i < 8; ++i) { a[i] = lo[i]; a[i + 8] = hv[i]; }
    return a;
}

// B fragment (32x16 f16, KxN). Source stored as (out-col j, k) row-major
// (== original torch weight layout), ldb in elements.
// ISA wave32 B layout: lane l (n = l&15, hi = l>>4): elems 0..7 -> k = k0 + hi*8 + i, elems 8..15 -> k = k0 + 16 + hi*8 + (i-8)
__device__ inline v16h load_frag_b(const _Float16* __restrict__ base, int ldb,
                                   int n0, int k0) {
    int lane = threadIdx.x & 31;
    int r = lane & 15, hi = (lane >> 4) & 1;
    const _Float16* p = base + (size_t)(n0 + r) * ldb + k0 + hi * 8;
    h8 lo = *(const h8*)p;
    h8 hv = *(const h8*)(p + 16);
    v16h b;
#pragma unroll
    for (int i = 0; i < 8; ++i) { b[i] = lo[i]; b[i + 8] = hv[i]; }
    return b;
}

// ---------------- register-blocked WMMA GEMM (2x2 tiles / wave) ----------------
// out(128 x ncols, f32) = A(128xK f16 row-major) @ B(ncols x K f16 row-major)^T + bias
// Each wave owns a 32x32 output block: 4 accumulators, 2 A-frags + 2 B-frags per
// k-step -> 4 WMMAs per 8 b128 loads (2 loads/WMMA), halving L2 traffic vs 1 tile/wave.
// Launch: total waves = 4 * (ncols/32); blockDim = 256 (8 waves).
__global__ void gemm_wmma_blk(const _Float16* __restrict__ A, int lda,
                              const _Float16* __restrict__ B, int ldb,
                              const float* __restrict__ bias,
                              float* __restrict__ out, int ldo,
                              int K, int nsn /* = ncols/32 */) {
    int wid = blockIdx.x * (blockDim.x >> 5) + (threadIdx.x >> 5);
    int ms = wid / nsn, ns = wid % nsn;
    int m0 = ms * 32, n0 = ns * 32;

    v8f c00 = {}, c01 = {}, c10 = {}, c11 = {};
    for (int k0 = 0; k0 < K; k0 += 32) {
        v16h a0 = load_frag_a(A, lda, m0,      k0);
        v16h a1 = load_frag_a(A, lda, m0 + 16, k0);
        v16h b0 = load_frag_b(B, ldb, n0,      k0);
        v16h b1 = load_frag_b(B, ldb, n0 + 16, k0);
        c00 = __builtin_amdgcn_wmma_f32_16x16x32_f16(false, a0, false, b0, (short)0, c00, false, false);
        c01 = __builtin_amdgcn_wmma_f32_16x16x32_f16(false, a0, false, b1, (short)0, c01, false, false);
        c10 = __builtin_amdgcn_wmma_f32_16x16x32_f16(false, a1, false, b0, (short)0, c10, false, false);
        c11 = __builtin_amdgcn_wmma_f32_16x16x32_f16(false, a1, false, b1, (short)0, c11, false, false);
    }

    int lane = threadIdx.x & 31;
    int hi = (lane >> 4) & 1;
    int nc0 = n0 + (lane & 15);
    int nc1 = nc0 + 16;
    float bv0 = bias ? bias[nc0] : 0.f;
    float bv1 = bias ? bias[nc1] : 0.f;
#pragma unroll
    for (int j = 0; j < 8; ++j) {
        int mr0 = m0 + hi * 8 + j;
        int mr1 = mr0 + 16;
        out[(size_t)mr0 * ldo + nc0] = c00[j] + bv0;
        out[(size_t)mr0 * ldo + nc1] = c01[j] + bv1;
        out[(size_t)mr1 * ldo + nc0] = c10[j] + bv0;
        out[(size_t)mr1 * ldo + nc1] = c11[j] + bv1;
    }
}

// ---------------- attention + embedding ----------------
// one block (256 thr) per batch row n: scores = (h1[n] . key[n,l]) * scale,
// softmax over L, ctx = attn . value; writes x0[n] = [emb | ctx | (h kept)]
__global__ void attn_kernel(const float* __restrict__ key,
                            const float* __restrict__ value,
                            const float* __restrict__ h1,
                            const int*   __restrict__ word,
                            const _Float16* __restrict__ embh,
                            _Float16* __restrict__ x0, int t) {
    __shared__ float qs[HH];
    __shared__ float sc[LL];
    __shared__ float s_inv;
    int n = blockIdx.x;
    int tid = threadIdx.x;
    int wid = tid >> 5, lane = tid & 31;
    const float scale = 0.044194173824159216f;  // 1/sqrt(512)

    for (int i = tid; i < HH; i += 256) qs[i] = h1[(size_t)n * HH + i];
    __syncthreads();

    // scores: each wave owns l = wid, wid+8, ...
    for (int l = wid; l < LL; l += 8) {
        const float* kr = key + ((size_t)n * LL + l) * HH;
        float acc = 0.f;
        for (int h = lane; h < HH; h += 32) acc += kr[h] * qs[h];
#pragma unroll
        for (int off = 16; off; off >>= 1) acc += __shfl_xor(acc, off);
        if (lane == 0) sc[l] = acc * scale;
    }
    __syncthreads();

    // softmax (unnormalized exp kept in sc, 1/sum in s_inv) — wave 0
    if (wid == 0) {
        float m = -1e30f;
        for (int i = lane; i < LL; i += 32) m = fmaxf(m, sc[i]);
#pragma unroll
        for (int off = 16; off; off >>= 1) m = fmaxf(m, __shfl_xor(m, off));
        float s = 0.f;
        for (int i = lane; i < LL; i += 32) {
            float e = __expf(sc[i] - m);
            sc[i] = e;
            s += e;
        }
#pragma unroll
        for (int off = 16; off; off >>= 1) s += __shfl_xor(s, off);
        if (lane == 0) s_inv = 1.f / s;
    }
    __syncthreads();

    // ctx + emb -> x0 (f16)
    float inv = s_inv;
    for (int h = tid; h < HH; h += 256) {
        float acc = 0.f;
        const float* vb = value + (size_t)n * LL * HH + h;
        for (int l = 0; l < LL; ++l) acc += sc[l] * vb[(size_t)l * HH];
        x0[(size_t)n * K0 + HH + h] = (_Float16)(acc * inv);
    }
    int tok = (t == 0) ? 0 : word[n * TT + (t - 1)];
    for (int h = tid; h < HH; h += 256)
        x0[(size_t)n * K0 + h] = embh[(size_t)tok * HH + h];
}

// ---------------- LSTM cell elementwise ----------------
__global__ void lstm_cell(const float* __restrict__ gates,
                          float* __restrict__ c, float* __restrict__ h,
                          _Float16* __restrict__ cp1, int ld1, int off1,
                          _Float16* __restrict__ cp2, int ld2, int off2) {
    int idx = blockIdx.x * 256 + threadIdx.x;
    if (idx >= NB * HH) return;
    int n = idx >> 9, u = idx & (HH - 1);
    const float* g = gates + (size_t)n * G4H;
    float gi = g[u], gf = g[HH + u], gg = g[2 * HH + u], go = g[3 * HH + u];
    float si = 1.f / (1.f + __expf(-gi));
    float sf = 1.f / (1.f + __expf(-gf));
    float so = 1.f / (1.f + __expf(-go));
    float cn = sf * c[idx] + si * tanhf(gg);
    float hn = so * tanhf(cn);
    c[idx] = cn;
    h[idx] = hn;
    cp1[(size_t)n * ld1 + off1 + u] = (_Float16)hn;
    if (cp2) cp2[(size_t)n * ld2 + off2 + u] = (_Float16)hn;
}

// ---------------- per-row softmax over V=64 -> probs ----------------
__global__ void softmax_probs(const float* __restrict__ logits,
                              float* __restrict__ probs, int t) {
    __shared__ float red[VV];
    int n = blockIdx.x, v = threadIdx.x;
    float x = logits[n * VV + v];
    red[v] = x; __syncthreads();
    for (int s = 32; s > 0; s >>= 1) {
        if (v < s) red[v] = fmaxf(red[v], red[v + s]);
        __syncthreads();
    }
    float m = red[0]; __syncthreads();
    float e = __expf(x - m);
    red[v] = e; __syncthreads();
    for (int s = 32; s > 0; s >>= 1) {
        if (v < s) red[v] += red[v + s];
        __syncthreads();
    }
    probs[((size_t)n * TT + t) * VV + v] = e / red[0];
}

// ---------------- setup / pack kernels ----------------
__global__ void pack_w0(const float* __restrict__ Wih, const float* __restrict__ Whh,
                        _Float16* __restrict__ out) {  // 2048 x 1536
    int idx = blockIdx.x * 256 + threadIdx.x;
    if (idx >= G4H * K0) return;
    int j = idx / K0, k = idx % K0;
    float v = (k < 1024) ? Wih[(size_t)j * 1024 + k] : Whh[(size_t)j * 512 + (k - 1024)];
    out[idx] = (_Float16)v;
}
__global__ void pack_w1(const float* __restrict__ Wih, const float* __restrict__ Whh,
                        _Float16* __restrict__ out) {  // 2048 x 1024
    int idx = blockIdx.x * 256 + threadIdx.x;
    if (idx >= G4H * K1) return;
    int j = idx / K1, k = idx % K1;
    float v = (k < 512) ? Wih[(size_t)j * 512 + k] : Whh[(size_t)j * 512 + (k - 512)];
    out[idx] = (_Float16)v;
}
__global__ void f32_to_f16(const float* __restrict__ in, _Float16* __restrict__ out, int n) {
    int idx = blockIdx.x * 256 + threadIdx.x;
    if (idx < n) out[idx] = (_Float16)in[idx];
}
__global__ void add_bias(const float* __restrict__ a, const float* __restrict__ b,
                         float* __restrict__ out, int n) {
    int idx = blockIdx.x * 256 + threadIdx.x;
    if (idx < n) out[idx] = a[idx] + b[idx];
}
__global__ void init_state(const float* __restrict__ eh, const float* __restrict__ ec,
                           float* __restrict__ h0, float* __restrict__ c0,
                           float* __restrict__ h1, float* __restrict__ c1,
                           _Float16* __restrict__ x0, _Float16* __restrict__ x1) {
    int idx = blockIdx.x * 256 + threadIdx.x;
    if (idx >= NB * HH) return;
    int n = idx >> 9, u = idx & (HH - 1);
    float a0 = eh[idx], a1 = eh[NB * HH + idx];
    h0[idx] = a0; h1[idx] = a1;
    c0[idx] = ec[idx]; c1[idx] = ec[NB * HH + idx];
    x0[(size_t)n * K0 + 1024 + u] = (_Float16)a0;  // h0_prev slot in x0
    x1[(size_t)n * K1 + 512 + u]  = (_Float16)a1;  // h1_prev slot in x1
}
__global__ void final_copy(const float* __restrict__ h0, const float* __restrict__ h1,
                           const float* __restrict__ c0, const float* __restrict__ c1,
                           float* __restrict__ out) {
    int idx = blockIdx.x * 256 + threadIdx.x;
    if (idx >= NB * HH) return;
    const size_t base = (size_t)NB * TT * VV;  // 163840
    out[base + idx]               = h0[idx];
    out[base + 65536 + idx]       = h1[idx];
    out[base + 131072 + idx]      = c0[idx];
    out[base + 196608 + idx]      = c1[idx];
}

// ---------------- launch ----------------
extern "C" void kernel_launch(void* const* d_in, const int* in_sizes, int n_in,
                              void* d_out, int out_size, void* d_ws, size_t ws_size,
                              hipStream_t stream) {
    const float* key   = (const float*)d_in[0];
    const float* value = (const float*)d_in[1];
    const float* eh    = (const float*)d_in[2];
    const float* ec    = (const float*)d_in[3];
    const int*   word  = (const int*)  d_in[4];
    const float* emb   = (const float*)d_in[5];
    const float* Wih0  = (const float*)d_in[6];
    const float* Whh0  = (const float*)d_in[7];
    const float* bih0  = (const float*)d_in[8];
    const float* bhh0  = (const float*)d_in[9];
    const float* Wih1  = (const float*)d_in[10];
    const float* Whh1  = (const float*)d_in[11];
    const float* bih1  = (const float*)d_in[12];
    const float* bhh1  = (const float*)d_in[13];
    const float* fcw   = (const float*)d_in[14];
    const float* fcb   = (const float*)d_in[15];
    float* out = (float*)d_out;

    // ---- carve scratch ----
    size_t off = 0;
    auto carve = [&](size_t bytes) -> void* {
        void* p = (char*)d_ws + off;
        off += (bytes + 255) & ~(size_t)255;
        return p;
    };
    _Float16* Wc0h  = (_Float16*)carve((size_t)G4H * K0 * 2);
    _Float16* Wc1h  = (_Float16*)carve((size_t)G4H * K1 * 2);
    _Float16* fcwh  = (_Float16*)carve((size_t)VV * HH * 2);
    _Float16* embh  = (_Float16*)carve((size_t)VV * HH * 2);
    float*    bc0   = (float*)   carve((size_t)G4H * 4);
    float*    bc1   = (float*)   carve((size_t)G4H * 4);
    _Float16* x0    = (_Float16*)carve((size_t)NB * K0 * 2);
    _Float16* x1    = (_Float16*)carve((size_t)NB * K1 * 2);
    float*    gates = (float*)   carve((size_t)NB * G4H * 4);
    float*    h0    = (float*)   carve((size_t)NB * HH * 4);
    float*    c0    = (float*)   carve((size_t)NB * HH * 4);
    float*    h1    = (float*)   carve((size_t)NB * HH * 4);
    float*    c1    = (float*)   carve((size_t)NB * HH * 4);
    float*    logit = (float*)   carve((size_t)NB * VV * 4);
    (void)ws_size; (void)n_in; (void)in_sizes; (void)out_size;

    // ---- one-time packs (re-run every call: deterministic) ----
    pack_w0<<<(G4H * K0 + 255) / 256, 256, 0, stream>>>(Wih0, Whh0, Wc0h);
    pack_w1<<<(G4H * K1 + 255) / 256, 256, 0, stream>>>(Wih1, Whh1, Wc1h);
    f32_to_f16<<<(VV * HH + 255) / 256, 256, 0, stream>>>(fcw, fcwh, VV * HH);
    f32_to_f16<<<(VV * HH + 255) / 256, 256, 0, stream>>>(emb, embh, VV * HH);
    add_bias<<<(G4H + 255) / 256, 256, 0, stream>>>(bih0, bhh0, bc0, G4H);
    add_bias<<<(G4H + 255) / 256, 256, 0, stream>>>(bih1, bhh1, bc1, G4H);
    init_state<<<(NB * HH + 255) / 256, 256, 0, stream>>>(eh, ec, h0, c0, h1, c1, x0, x1);

    // ---- sequential decode ----
    for (int t = 0; t < TT; ++t) {
        // attention + embedding -> x0 = [emb | ctx | h0_prev(already there)]
        attn_kernel<<<NB, 256, 0, stream>>>(key, value, h1, word, embh, x0, t);
        // gates0 = x0 @ Wc0^T + bc0 (128 x 2048, K=1536): 4x64 block-strips = 256 waves
        gemm_wmma_blk<<<32, 256, 0, stream>>>(x0, K0, Wc0h, K0, bc0, gates, G4H, K0, 64);
        // cell0: updates h0/c0; writes h0 f16 into x1[:,0:512] and x0[:,1024:1536]
        lstm_cell<<<(NB * HH + 255) / 256, 256, 0, stream>>>(
            gates, c0, h0, x1, K1, 0, x0, K0, 1024);
        // gates1 = x1 @ Wc1^T + bc1 (128 x 2048, K=1024): 256 waves
        gemm_wmma_blk<<<32, 256, 0, stream>>>(x1, K1, Wc1h, K1, bc1, gates, G4H, K1, 64);
        // cell1: updates h1/c1; writes h1 f16 into x1[:,512:1024]
        lstm_cell<<<(NB * HH + 255) / 256, 256, 0, stream>>>(
            gates, c1, h1, x1, K1, 512, (_Float16*)nullptr, 0, 0);
        // logits = h1 @ fc_w^T + fc_b (128 x 64, K=512): 4x2 strips = 8 waves, 1 block
        gemm_wmma_blk<<<1, 256, 0, stream>>>(x1 + 512, K1, fcwh, HH, fcb, logit, VV, HH, 2);
        // probs[:, t, :]
        softmax_probs<<<NB, VV, 0, stream>>>(logit, out, t);
    }

    // ---- outputs: h_fin (2,N,H), c_fin (2,N,H) after probs ----
    final_copy<<<(NB * HH + 255) / 256, 256, 0, stream>>>(h0, h1, c0, c1, out);
}
